// MoEGateTTNN_71803263255219
// MI455X (gfx1250) — compile-verified, matrix operand order checked by reading, not run on hardware
//
#include <hip/hip_runtime.h>
#include <hip/hip_bf16.h>

// ---------------- problem constants ----------------
#define T_TOK   8192
#define H_DIM   7168
#define E_EXP   256
#define N_GROUP 8
#define GSIZE   32          // E / N_GROUP
#define TOPK_G  4
#define TOP_K   8
#define KCHUNK  512         // K staging chunk (divides 7168: 14 chunks)
#define MBLK    32          // tokens per block (2 M-tiles per wave)

typedef __attribute__((ext_vector_type(16))) __bf16 v16bf;
typedef __attribute__((ext_vector_type(8)))  float  v8f;

union Frag { uint4 q[2]; v16bf v; };

// round-to-nearest-even float -> bf16 bits
__device__ __forceinline__ unsigned short f2bf(float f) {
    unsigned u = __float_as_uint(f);
    unsigned r = u + 0x7FFFu + ((u >> 16) & 1u);
    return (unsigned short)(r >> 16);
}
__device__ __forceinline__ float bf2f(unsigned short h) {
    return __uint_as_float(((unsigned)h) << 16);
}

#define WMMA_BF16(A_, B_, C_) \
    __builtin_amdgcn_wmma_f32_16x16x32_bf16(false, (A_), false, (B_), (short)0, (C_), false, false)

// ---------------------------------------------------------------------------
// Kernel 1: transpose + hi/lo-split weight  [H,E] f32  ->  Wt_hi/Wt_lo [E,H] bf16
// ---------------------------------------------------------------------------
__global__ __launch_bounds__(256)
void wsplit_kernel(const float* __restrict__ W,
                   unsigned short* __restrict__ wt_hi,
                   unsigned short* __restrict__ wt_lo) {
    __shared__ float tile[32][33];
    const int h0 = blockIdx.x * 32;
    const int e0 = blockIdx.y * 32;
    const int tid = threadIdx.x;
    const int j  = tid & 31;   // e within tile (coalesced read)
    const int i0 = tid >> 5;   // 8 rows per pass
#pragma unroll
    for (int r = 0; r < 4; ++r) {
        int row = i0 + r * 8;
        tile[row][j] = W[(size_t)(h0 + row) * E_EXP + e0 + j];
    }
    __syncthreads();
    const int hh = tid & 31;   // h within tile (coalesced write)
    const int ee0 = tid >> 5;
#pragma unroll
    for (int r = 0; r < 4; ++r) {
        int ee = ee0 + r * 8;
        float x = tile[hh][ee];
        unsigned short hi = f2bf(x);
        unsigned short lo = f2bf(x - bf2f(hi));
        size_t dst = (size_t)(e0 + ee) * H_DIM + h0 + hh;
        wt_hi[dst] = hi;
        wt_lo[dst] = lo;
    }
}

// ---------------------------------------------------------------------------
// Kernel 2: fused gate. One block = 32 tokens x 256 experts. 8 waves.
// Wave tile: 2 M-tiles x 2 N-tiles, compensated bf16 (hi*hi + hi*lo + lo*hi).
// ---------------------------------------------------------------------------
__global__ __launch_bounds__(256)
void moe_gate_kernel(const float* __restrict__ A,              // [T,H]
                     const unsigned short* __restrict__ wt_hi, // [E,H] bf16
                     const unsigned short* __restrict__ wt_lo, // [E,H] bf16
                     const float* __restrict__ bias,           // [E]
                     float* __restrict__ out) {                // [T*8 idx | T*8 w]
    __shared__ __align__(16) unsigned short lds_hi[MBLK * KCHUNK]; // 32 KB
    __shared__ __align__(16) unsigned short lds_lo[MBLK * KCHUNK]; // 32 KB
    __shared__ float lds_corr[MBLK][E_EXP];                        // 32 KB
    __shared__ float lds_scr [MBLK][E_EXP];                        // 32 KB

    const int tid  = threadIdx.x;
    const int wave = tid >> 5;
    const int lane = tid & 31;
    const int t0   = blockIdx.x * MBLK;

    const int n0    = wave * 32;      // this wave: experts [n0, n0+32)
    const int lmod  = lane & 15;
    const int lhalf = lane >> 4;      // 0 or 1
    const int koff8 = lhalf << 3;     // per-half-wave K sub-offset (A/B frag layout)

    v8f c00 = {0.f,0.f,0.f,0.f,0.f,0.f,0.f,0.f};
    v8f c01 = c00, c10 = c00, c11 = c00;

    // per-lane 32-bit byte offsets into the bf16 weight planes (max 3.67 MB)
    const char* whB = (const char*)wt_hi;
    const char* wlB = (const char*)wt_lo;
    const int bo0 = ((n0 + lmod)      * H_DIM + koff8) * 2;
    const int bo1 = ((n0 + 16 + lmod) * H_DIM + koff8) * 2;

    // LDS A-fragment element bases for the two M-tiles
    const int ab0 = lmod * KCHUNK + koff8;
    const int ab1 = ab0 + 16 * KCHUNK;

    // cooperative-load mapping: 8 threads per token row, 16 float4 each
    const int mrow = tid >> 3;        // 0..31
    const int sub  = tid & 7;         // 0..7

    for (int kc = 0; kc < H_DIM; kc += KCHUNK) {
        __syncthreads();   // previous chunk's LDS reads done
        // ---- stage A chunk: [32 tokens][KCHUNK] f32 -> hi/lo bf16 in LDS ----
        const float4* rowp =
            (const float4*)(A + (size_t)(t0 + mrow) * H_DIM + kc);
#pragma unroll
        for (int i = 0; i < 16; ++i) {
            int f4 = sub + i * 8;                  // float4 index 0..127
            float4 f = rowp[f4];
            unsigned short h0 = f2bf(f.x), h1 = f2bf(f.y),
                           h2 = f2bf(f.z), h3 = f2bf(f.w);
            unsigned short l0 = f2bf(f.x - bf2f(h0)), l1 = f2bf(f.y - bf2f(h1)),
                           l2 = f2bf(f.z - bf2f(h2)), l3 = f2bf(f.w - bf2f(h3));
            uint2 hp, lp;
            hp.x = (unsigned)h0 | ((unsigned)h1 << 16);
            hp.y = (unsigned)h2 | ((unsigned)h3 << 16);
            lp.x = (unsigned)l0 | ((unsigned)l1 << 16);
            lp.y = (unsigned)l2 | ((unsigned)l3 << 16);
            int k4 = f4 * 4;
            *(uint2*)&lds_hi[mrow * KCHUNK + k4] = hp;
            *(uint2*)&lds_lo[mrow * KCHUNK + k4] = lp;
        }
        __syncthreads();

        // ---- inner WMMA loop over this chunk ----
#pragma unroll 2
        for (int kk = 0; kk < KCHUNK; kk += 32) {
            // A fragments for both M-tiles (hi/lo), from LDS
            Frag ahi0, alo0, ahi1, alo1;
            ahi0.q[0] = *(const uint4*)&lds_hi[ab0 + kk];
            ahi0.q[1] = *(const uint4*)&lds_hi[ab0 + kk + 16];
            alo0.q[0] = *(const uint4*)&lds_lo[ab0 + kk];
            alo0.q[1] = *(const uint4*)&lds_lo[ab0 + kk + 16];
            ahi1.q[0] = *(const uint4*)&lds_hi[ab1 + kk];
            ahi1.q[1] = *(const uint4*)&lds_hi[ab1 + kk + 16];
            alo1.q[0] = *(const uint4*)&lds_lo[ab1 + kk];
            alo1.q[1] = *(const uint4*)&lds_lo[ab1 + kk + 16];

            // B fragments for both N-tiles (hi/lo), 32-bit offset addressing
            const int kb = (kc + kk) * 2;
            Frag bh0, bl0, bh1, bl1;
            bh0.q[0] = *(const uint4*)(whB + (bo0 + kb));
            bh0.q[1] = *(const uint4*)(whB + (bo0 + kb) + 32);
            bl0.q[0] = *(const uint4*)(wlB + (bo0 + kb));
            bl0.q[1] = *(const uint4*)(wlB + (bo0 + kb) + 32);
            bh1.q[0] = *(const uint4*)(whB + (bo1 + kb));
            bh1.q[1] = *(const uint4*)(whB + (bo1 + kb) + 32);
            bl1.q[0] = *(const uint4*)(wlB + (bo1 + kb));
            bl1.q[1] = *(const uint4*)(wlB + (bo1 + kb) + 32);

            // 12 chained WMMAs: compensated bf16 per (M-tile, N-tile)
            c00 = WMMA_BF16(ahi0.v, bh0.v, c00);
            c00 = WMMA_BF16(ahi0.v, bl0.v, c00);
            c00 = WMMA_BF16(alo0.v, bh0.v, c00);
            c01 = WMMA_BF16(ahi0.v, bh1.v, c01);
            c01 = WMMA_BF16(ahi0.v, bl1.v, c01);
            c01 = WMMA_BF16(alo0.v, bh1.v, c01);
            c10 = WMMA_BF16(ahi1.v, bh0.v, c10);
            c10 = WMMA_BF16(ahi1.v, bl0.v, c10);
            c10 = WMMA_BF16(alo1.v, bh0.v, c10);
            c11 = WMMA_BF16(ahi1.v, bh1.v, c11);
            c11 = WMMA_BF16(ahi1.v, bl1.v, c11);
            c11 = WMMA_BF16(alo1.v, bh1.v, c11);
        }
    }

    // ---- epilogue: sigmoid + bias into LDS score arrays ----
    const float b0 = bias[n0 + lmod];
    const float b1 = bias[n0 + 16 + lmod];
#pragma unroll
    for (int r = 0; r < 8; ++r) {
        const int mr = r + (lhalf << 3);          // row within a 16-token tile
        const int n  = n0 + lmod;
        float s;
        s = 1.0f / (1.0f + __expf(-c00[r]));
        lds_scr[mr][n] = s;            lds_corr[mr][n] = s + b0;
        s = 1.0f / (1.0f + __expf(-c01[r]));
        lds_scr[mr][n + 16] = s;       lds_corr[mr][n + 16] = s + b1;
        s = 1.0f / (1.0f + __expf(-c10[r]));
        lds_scr[mr + 16][n] = s;       lds_corr[mr + 16][n] = s + b0;
        s = 1.0f / (1.0f + __expf(-c11[r]));
        lds_scr[mr + 16][n + 16] = s;  lds_corr[mr + 16][n + 16] = s + b1;
    }
    __syncthreads();

    // ---- grouped top-k: one lane per token ----
    if (tid < MBLK) {
        const int m = tid;
        float* corr = lds_corr[m];
        float* scr  = lds_scr[m];
        const float NEG = -__builtin_inff();

        // per-group score = sum of top-2 corrected scores
        float gs[N_GROUP];
#pragma unroll
        for (int g = 0; g < N_GROUP; ++g) {
            float t1 = NEG, t2 = NEG;
            for (int j = 0; j < GSIZE; ++j) {
                float v = corr[g * GSIZE + j];
                if (v > t1)      { t2 = t1; t1 = v; }
                else if (v > t2) { t2 = v; }
            }
            gs[g] = t1 + t2;
        }
        // top-4 groups
        unsigned chosen = 0;
#pragma unroll
        for (int p = 0; p < TOPK_G; ++p) {
            float best = NEG; int bi = 0;
#pragma unroll
            for (int g = 0; g < N_GROUP; ++g)
                if (!((chosen >> g) & 1u) && gs[g] > best) { best = gs[g]; bi = g; }
            chosen |= 1u << bi;
        }
        // top-8 experts within chosen groups (mark-taken with -inf)
        int   id[TOP_K];
        float wv[TOP_K];
        float wsum = 0.0f;
#pragma unroll
        for (int p = 0; p < TOP_K; ++p) {
            float best = NEG; int bi = 0;
            for (int g = 0; g < N_GROUP; ++g) {
                if (!((chosen >> g) & 1u)) continue;
                for (int j = 0; j < GSIZE; ++j) {
                    float v = corr[g * GSIZE + j];
                    if (v > best) { best = v; bi = g * GSIZE + j; }
                }
            }
            corr[bi] = NEG;
            float s = scr[bi];   // weights come from ORIGINAL sigmoid scores
            id[p] = bi; wv[p] = s; wsum += s;
        }
        float scale = 2.5f / (wsum + 1e-20f);
        const int t = t0 + m;
        float* out_idx = out;
        float* out_w   = out + (size_t)T_TOK * TOP_K;
#pragma unroll
        for (int p = 0; p < TOP_K; ++p) {
            out_idx[t * TOP_K + p] = (float)id[p];
            out_w  [t * TOP_K + p] = wv[p] * scale;
        }
    }
}

// ---------------------------------------------------------------------------
extern "C" void kernel_launch(void* const* d_in, const int* in_sizes, int n_in,
                              void* d_out, int out_size, void* d_ws, size_t ws_size,
                              hipStream_t stream) {
    const float* A    = (const float*)d_in[0];   // [T,H]
    const float* W    = (const float*)d_in[1];   // [H,E]
    const float* bias = (const float*)d_in[2];   // [E]
    float* out = (float*)d_out;

    unsigned short* wt_hi = (unsigned short*)d_ws;
    unsigned short* wt_lo = wt_hi + (size_t)E_EXP * H_DIM;

    dim3 gprep(H_DIM / 32, E_EXP / 32);
    wsplit_kernel<<<gprep, 256, 0, stream>>>(W, wt_hi, wt_lo);

    moe_gate_kernel<<<T_TOK / MBLK, 256, 0, stream>>>(A, wt_hi, wt_lo, bias, out);
}
